// SimpleBlock_82480551952816
// MI455X (gfx1250) — matrix operand order repaired and checked by reading, not compiled, not run on hardware
//
#include <hip/hip_runtime.h>
#include <hip/hip_bf16.h>

// ---------------------------------------------------------------------------
// KPConv + BatchNorm1d(train fwd) + LeakyReLU(0.1) for MI455X (gfx1250).
// GEMM stage uses v_wmma_f32_16x16x32_f16 (wave32 WMMA, f32 accumulate).
// 32 queries per WG: each packed-B fragment feeds two WMMAs (M-tiles),
// halving L2 traffic for the weight matrix.
// ---------------------------------------------------------------------------

typedef __attribute__((ext_vector_type(16))) _Float16 v16h;
typedef __attribute__((ext_vector_type(8)))  _Float16 v8h;
typedef __attribute__((ext_vector_type(4)))  _Float16 v4h;
typedef __attribute__((ext_vector_type(8)))  float    v8f;
typedef __attribute__((ext_vector_type(4)))  float    v4f;

#define NPTS   100000      // support points
#define MQ     100000      // query points
#define NBR    16          // neighbors per query
#define NKP    15          // kernel points
#define CIN    64
#define COUTC  128
#define KDIM   (NKP * CIN)     // 960 = GEMM K dimension
#define ROWH   (KDIM + 8)      // padded LDS A row stride (484 dwords: bank-spread)
#define KTILES (KDIM / 32)     // 30 WMMA K-tiles
#define NTILES (COUTC / 16)    // 8 WMMA N-tiles
#define MQWG   32              // queries per workgroup (2 WMMA M-tiles)
#define INV_KP_EXTENT 10.0f    // 1 / 0.1
#define BN_EPS 1e-5f
#define NEG_SLOPE 0.1f
#define RED_WGS 250            // bn_reduce workgroups (400 rows each)

union Frag16 { v16h v; v8h h[2]; };

// ---------------------------------------------------------------------------
// K0: repack weight [960 x 128] fp32 -> fp16 in WMMA B-fragment order.
// Packed layout: for kt in [0,30), nt in [0,8), lane in [0,32): 16 halves,
// half i of lane l = W[kt*32 + (l/16)*16 + i][nt*16 + (l&15)]
// (contiguous k order chosen here, A uses the same; B 32x16: lanes 0-15 hold K=0..15, lanes 16-31 hold K=16..31,
//  lane%16 = column N, VGPR/half index = K within the half-wave group.)
// ---------------------------------------------------------------------------
__global__ void __launch_bounds__(256) pack_weight(const float* __restrict__ w,
                                                   _Float16* __restrict__ pw) {
    int tid = blockIdx.x * 256 + threadIdx.x;
    if (tid >= KDIM * COUTC) return;
    int i  = tid & 15;
    int l  = (tid >> 4) & 31;
    int nt = (tid >> 9) & 7;
    int kt = tid >> 12;
    int kg = kt * 32 + ((l >> 4) << 4) + i;   // GEMM K index (= kp*64 + cin)
    int o  = nt * 16 + (l & 15);              // output channel
    pw[tid] = (_Float16)w[kg * COUTC + o];
}

// ---------------------------------------------------------------------------
// K1: per-WG = 32 queries. Build A tile [32 x 960] in LDS (f16, padded rows),
// then 8 waves x (30 kt x 2 M-tiles) WMMAs -> pre-BN [32 x 128] into d_out.
// ---------------------------------------------------------------------------
__global__ void __launch_bounds__(256) kpconv_wmma(
    const float* __restrict__ qpts, const float* __restrict__ spts,
    const float* __restrict__ feats, const float* __restrict__ kpts,
    const int* __restrict__ edges, const _Float16* __restrict__ pw,
    float* __restrict__ out)
{
    __shared__ float    sKP[NKP * 3];
    __shared__ float    sW[MQWG * NBR * NKP];   // influence weights (30720 B)
    __shared__ int      sIdx[MQWG * NBR];       // clamped neighbor indices
    __shared__ _Float16 sA[MQWG * ROWH];        // A tile, padded rows (61952 B)

    const int t = threadIdx.x;
    const int qbase = blockIdx.x * MQWG;        // 3125 * 32 == 100000 exactly

    if (t < NKP * 3) sKP[t] = kpts[t];
    __syncthreads();

    // ---- phase 1: influence weights, 2 (query, neighbor) pairs / thread ----
#pragma unroll
    for (int h = 0; h < 2; ++h) {
        const int p = t + h * 256;
        const int q = p >> 4, n = p & 15;
        const int mq = qbase + q;
        int idx = edges[mq * NBR + n];
        const bool shadow = (idx >= NPTS);      // shadow/padding slot
        const int idxc = shadow ? 0 : idx;
        sIdx[q * NBR + n] = idxc;
        const float qx = qpts[mq * 3 + 0];
        const float qy = qpts[mq * 3 + 1];
        const float qz = qpts[mq * 3 + 2];
        const float dx = spts[idxc * 3 + 0] - qx;
        const float dy = spts[idxc * 3 + 1] - qy;
        const float dz = spts[idxc * 3 + 2] - qz;
#pragma unroll
        for (int k = 0; k < NKP; ++k) {
            const float ex = dx - sKP[k * 3 + 0];
            const float ey = dy - sKP[k * 3 + 1];
            const float ez = dz - sKP[k * 3 + 2];
            const float d  = sqrtf(ex * ex + ey * ey + ez * ez);
            float wv = fmaxf(1.0f - d * INV_KP_EXTENT, 0.0f);
            sW[(q * NBR + n) * NKP + k] = shadow ? 0.0f : wv;
        }
    }
    __syncthreads();

    // ---- phase 2: A[m, k*64+c] = sum_n w[m,n,k] * f[idx(m,n), c] ----
    // 4-wide float vectors -> packed f32 FMAs; two 16-query halves in turn.
#pragma unroll
    for (int h = 0; h < 2; ++h) {
        const int q = (t >> 4) + h * 16;
        const int g = t & 15;
        const int c0 = g * 4;
        v4f acc[NKP];
#pragma unroll
        for (int k = 0; k < NKP; ++k) acc[k] = (v4f)0.0f;

#pragma unroll 2
        for (int n = 0; n < NBR; ++n) {
            const int idx = sIdx[q * NBR + n];
            const v4f f = *(const v4f*)(feats + idx * CIN + c0);
            const float* wrow = &sW[(q * NBR + n) * NKP];
#pragma unroll
            for (int k = 0; k < NKP; ++k) {
                acc[k] += f * wrow[k];
            }
        }
#pragma unroll
        for (int k = 0; k < NKP; ++k) {
            v4h hh = { (_Float16)acc[k][0], (_Float16)acc[k][1],
                       (_Float16)acc[k][2], (_Float16)acc[k][3] };
            *(v4h*)&sA[q * ROWH + k * CIN + c0] = hh;
        }
    }
    __syncthreads();

    // ---- phase 3: GEMM [32 x 960] x [960 x 128], wave w -> COUT tile w ----
    // One B fragment per kt feeds both M-tiles (independent accumulators).
    {
        const int wvid = t >> 5;                // 0..7 N-tile
        const int l    = t & 31;
        const int mrow = l & 15;                // ISA A layout: lane%16 = M
        const int hi8  = (l >> 4) << 4;         // lanes>=16 start at K+16

        v8f c0 = {};
        v8f c1 = {};
        for (int kt = 0; kt < KTILES; ++kt) {
            const v16h b = *(const v16h*)(pw + ((kt * NTILES + wvid) * 32 + l) * 16);
            const _Float16* a0 = &sA[mrow * ROWH + kt * 32 + hi8];
            const _Float16* a1 = &sA[(mrow + 16) * ROWH + kt * 32 + hi8];
            Frag16 f0; f0.h[0] = *(const v8h*)a0; f0.h[1] = *(const v8h*)(a0 + 8);
            Frag16 f1; f1.h[0] = *(const v8h*)a1; f1.h[1] = *(const v8h*)(a1 + 8);
            c0 = __builtin_amdgcn_wmma_f32_16x16x32_f16(
                    false, f0.v, false, b, (short)0, c0, false, false);
            c1 = __builtin_amdgcn_wmma_f32_16x16x32_f16(
                    false, f1.v, false, b, (short)0, c1, false, false);
        }
        // C/D layout: VGPR r, lanes 0-15 -> M=r, lanes 16-31 -> M=r+8; N=lane%16
        const int mofs = (l >> 4) << 3;
        const int col  = wvid * 16 + (l & 15);
#pragma unroll
        for (int r = 0; r < 8; ++r) {
            out[(qbase + r + mofs) * COUTC + col] = c0[r];
            out[(qbase + 16 + r + mofs) * COUTC + col] = c1[r];
        }
    }
}

// ---------------------------------------------------------------------------
// K2: per-channel partial sum / sumsq over 400 rows per WG (no atomics).
// partials layout: [wg][0:128]=sum, [wg][128:256]=sumsq
// ---------------------------------------------------------------------------
__global__ void __launch_bounds__(256) bn_reduce(const float* __restrict__ x,
                                                 float* __restrict__ partials) {
    __shared__ float sS[256], sQ[256];
    const int t  = threadIdx.x;
    const int ch = t & 127, rg = t >> 7;
    const int r0 = blockIdx.x * (MQ / RED_WGS);          // 400 rows per WG
    float s = 0.0f, q = 0.0f;
    for (int r = r0 + rg; r < r0 + (MQ / RED_WGS); r += 2) {
        const float v = x[r * COUTC + ch];
        s += v; q += v * v;
    }
    sS[t] = s; sQ[t] = q;
    __syncthreads();
    if (t < 128)
        partials[blockIdx.x * 256 + t] = sS[t] + sS[t + 128];
    else
        partials[blockIdx.x * 256 + t] = sQ[t - 128] + sQ[t];
}

// ---------------------------------------------------------------------------
// K3: fold partials -> per-channel scale/shift (biased variance, like ref).
// ---------------------------------------------------------------------------
__global__ void bn_finalize(const float* __restrict__ partials,
                            const float* __restrict__ gamma,
                            const float* __restrict__ beta,
                            float* __restrict__ sb) {
    const int c = threadIdx.x;                  // 128 threads
    float s = 0.0f, q = 0.0f;
    for (int w = 0; w < RED_WGS; ++w) {
        s += partials[w * 256 + c];
        q += partials[w * 256 + 128 + c];
    }
    const float mean = s * (1.0f / (float)MQ);
    float var = q * (1.0f / (float)MQ) - mean * mean;
    var = fmaxf(var, 0.0f);
    const float inv = rsqrtf(var + BN_EPS);
    const float sc  = gamma[c] * inv;
    sb[c]       = sc;
    sb[128 + c] = beta[c] - mean * sc;
}

// ---------------------------------------------------------------------------
// K4: in-place y = leaky(x * scale + shift), float4 vectorized.
// ---------------------------------------------------------------------------
__global__ void __launch_bounds__(256) bn_act(float* __restrict__ x,
                                              const float* __restrict__ sb) {
    __shared__ float sc[128], sh[128];
    const int t = threadIdx.x;
    if (t < 128) { sc[t] = sb[t]; sh[t] = sb[128 + t]; }
    __syncthreads();
    const int e = (blockIdx.x * 256 + t) * 4;   // 12500*1024 == MQ*COUTC
    if (e >= MQ * COUTC) return;
    const int c0 = e & 127;
    float4 v = *(float4*)(x + e);
    float y0 = v.x * sc[c0 + 0] + sh[c0 + 0];
    float y1 = v.y * sc[c0 + 1] + sh[c0 + 1];
    float y2 = v.z * sc[c0 + 2] + sh[c0 + 2];
    float y3 = v.w * sc[c0 + 3] + sh[c0 + 3];
    v.x = (y0 >= 0.0f) ? y0 : NEG_SLOPE * y0;
    v.y = (y1 >= 0.0f) ? y1 : NEG_SLOPE * y1;
    v.z = (y2 >= 0.0f) ? y2 : NEG_SLOPE * y2;
    v.w = (y3 >= 0.0f) ? y3 : NEG_SLOPE * y3;
    *(float4*)(x + e) = v;
}

// ---------------------------------------------------------------------------
extern "C" void kernel_launch(void* const* d_in, const int* in_sizes, int n_in,
                              void* d_out, int out_size, void* d_ws, size_t ws_size,
                              hipStream_t stream) {
    (void)in_sizes; (void)n_in; (void)out_size; (void)ws_size;
    const float* qp = (const float*)d_in[0];    // query_points   [M,3]
    const float* sp = (const float*)d_in[1];    // support_points [N,3]
    const float* ft = (const float*)d_in[2];    // features       [N,64]
    const float* kp = (const float*)d_in[3];    // kernel_points  [15,3]
    const float* wt = (const float*)d_in[4];    // weight         [15,64,128]
    const float* ga = (const float*)d_in[5];    // gamma          [128]
    const float* be = (const float*)d_in[6];    // beta           [128]
    const int*   ei = (const int*)d_in[7];      // edge_indices   [M,16]
    float* out = (float*)d_out;                 // [M,128] f32

    // workspace layout
    _Float16* pw     = (_Float16*)d_ws;                              // 245760 B
    float* partials  = (float*)((char*)d_ws + (size_t)KDIM * COUTC * 2); // 256000 B
    float* sb        = partials + RED_WGS * 256;                     // 1024 B

    pack_weight<<<(KDIM * COUTC + 255) / 256, 256, 0, stream>>>(wt, pw);
    kpconv_wmma<<<MQ / MQWG, 256, 0, stream>>>(qp, sp, ft, kp, ei, pw, out);
    bn_reduce<<<RED_WGS, 256, 0, stream>>>(out, partials);
    bn_finalize<<<1, 128, 0, stream>>>(partials, ga, be, sb);
    bn_act<<<(MQ * COUTC) / 1024, 256, 0, stream>>>(out, sb);
}
